// MambaFormerBlock_63634235458203
// MI455X (gfx1250) — matrix operand order repaired and checked by reading, not compiled, not run on hardware
//
#include <hip/hip_runtime.h>
#include <hip/hip_bf16.h>

typedef __attribute__((ext_vector_type(8)))  _Float16     v8h;
typedef __attribute__((ext_vector_type(16))) _Float16     v16h;
typedef __attribute__((ext_vector_type(8)))  float        v8f;
typedef __attribute__((ext_vector_type(4)))  unsigned int v4u;
typedef __attribute__((ext_vector_type(8)))  int          v8i;
typedef __attribute__((ext_vector_type(4)))  int          v4i;

#define DIMC    64
#define D_INNER 128
#define D_STATE 16
#define D_CONVK 4
#define DT_RANK 4
#define HIDDEN  128
#define BB_     2
#define HH_     128
#define WW_     128
#define LL_     (HH_*WW_)        // 16384
#define MM_     (BB_*LL_)        // 32768
#define XDB_LD  48               // 36 cols padded to 48

// ---------------- LayerNorm (f32 in -> f16 out), NCHW-gather or row-major ----------------
__global__ __launch_bounds__(256)
void k_layernorm_f16(const float* __restrict__ in, int nchw,
                     const float* __restrict__ w, const float* __restrict__ b,
                     _Float16* __restrict__ out) {
  int r = blockIdx.x * 256 + threadIdx.x;
  if (r >= MM_) return;
  int bb = r >> 14, l = r & (LL_ - 1);
  float v[DIMC];
  float mu = 0.f;
  #pragma unroll
  for (int c = 0; c < DIMC; ++c) {
    float x = nchw ? in[((size_t)(bb * DIMC + c) << 14) + l]
                   : in[(size_t)r * DIMC + c];
    v[c] = x; mu += x;
  }
  mu *= (1.0f / DIMC);
  float var = 0.f;
  #pragma unroll
  for (int c = 0; c < DIMC; ++c) { float d = v[c] - mu; var += d * d; }
  var *= (1.0f / DIMC);
  float rstd = rsqrtf(var + 1e-5f);
  #pragma unroll
  for (int c = 0; c < DIMC; ++c)
    out[(size_t)r * DIMC + c] = (_Float16)((v[c] - mu) * rstd * w[c] + b[c]);
}

// ---------------- Templated WMMA GEMM: (M x K f16) @ (K x N f32 weights) ----------------
// A tile (128 x K f16, contiguous, lda==K) is DMA'd into LDS by the Tensor Data Mover;
// weights are staged transposed to WT[n][k] f16.  All loop trip counts compile-time.
// MODE 0: out[row*NPAD+col] = acc
// MODE 1: out[row*64+col]  = aux_nchw[(b*64+col)*L + l] + acc     (out_proj residual)
// MODE 2: dout[(b*64+col)*L + l] = aux_row[row*64+col] + acc      (final residual, NCHW)
template<int K, int N, int NPAD, bool WNK, int MODE>
__global__ __launch_bounds__(256)
void k_wmma_gemm(const _Float16* __restrict__ A,
                 const float* __restrict__ W,
                 float* __restrict__ out,
                 const float* __restrict__ aux, float* __restrict__ dout) {
  __shared__ _Float16 wt[NPAD * K];       // weights, N x K, f16
  __shared__ _Float16 atile[128 * K];     // activation tile via TDM
  const int tid = threadIdx.x;

  // TDM: async-copy this block's 128 x K f16 tile into LDS (one issuing lane).
  if (tid == 0) {
    unsigned long long ga =
        (unsigned long long)(A + (size_t)blockIdx.x * 128 * K);
    unsigned int lds = (unsigned int)(unsigned long long)(void*)atile;
    v4u g0 = { 1u,                                     // count=1 (valid D#)
               lds,                                    // lds_addr (bytes)
               (unsigned int)ga,                       // global_addr[31:0]
               ((unsigned int)(ga >> 32) & 0x01FFFFFFu) | (2u << 30) }; // addr[56:32] | type=2
    v8i g1 = { (int)(1u << 16),                        // data_size = 2 bytes
               (int)(((unsigned)K & 0xFFFFu) << 16),   // tensor_dim0[15:0] = K
               (int)(((unsigned)MM_ & 0xFFFFu) << 16), // tensor_dim1[15:0] = M rows
               (int)(((unsigned)MM_ >> 16) | ((unsigned)K << 16)), // dim1 hi | tile_dim0=K
               128,                                    // tile_dim1 = 128 rows
               K,                                      // tensor_dim0_stride = K
               0, 0 };
    v4i g2 = {0, 0, 0, 0}, g3 = {0, 0, 0, 0};
#if defined(__clang_major__) && __clang_major__ >= 23
    v8i g4 = {0, 0, 0, 0, 0, 0, 0, 0};
    __builtin_amdgcn_tensor_load_to_lds(g0, g1, g2, g3, g4, 0);
#else
    __builtin_amdgcn_tensor_load_to_lds(g0, g1, g2, g3, 0);
#endif
    __builtin_amdgcn_s_wait_tensorcnt(0);
  }

  // Stage weights transposed + f32->f16 (overlaps with TDM transfer).
  #pragma unroll 1
  for (int idx = tid; idx < NPAD * K; idx += 256) {
    int n = idx / K, k = idx - n * K;
    float val = 0.f;
    if (n < N) val = WNK ? W[n * K + k] : W[(size_t)k * N + n];
    wt[idx] = (_Float16)val;
  }
  __syncthreads();

  const int lane = tid & 31;
  const int wid  = tid >> 5;
  const int trow = wid * 16 + (lane & 15);           // row within LDS tile
  const int row0 = blockIdx.x * 128 + wid * 16;      // global row of this wave's tile
  constexpr int NK = K / 32;

  // A fragments: 16-bit A 16x32 layout (lanes 0-15: K=kb+{0..7,16..23}; lanes 16-31: +8)
  v16h afrag[NK];
  #pragma unroll
  for (int i = 0; i < NK; ++i) {
    int k0 = (i << 5) + ((lane >> 4) << 3);
    v8h lo = *(const v8h*)(atile + trow * K + k0);
    v8h hi = *(const v8h*)(atile + trow * K + k0 + 16);
    afrag[i] = __builtin_shufflevector(lo, hi, 0,1,2,3,4,5,6,7,8,9,10,11,12,13,14,15);
  }

  #pragma unroll 4
  for (int nt = 0; nt < NPAD / 16; ++nt) {
    v8f acc = {};
    const int ncol = (nt << 4) + (lane & 15);
    #pragma unroll
    for (int i = 0; i < NK; ++i) {
      // B fragment: lane holds column ncol, K=kb+{0..7,16..23} (lanes 16-31: +8)
      int koff = (i << 5) + ((lane >> 4) << 3);
      const _Float16* bp = &wt[ncol * K + koff];
      v8h blo = *(const v8h*)(bp);
      v8h bhi = *(const v8h*)(bp + 16);
      v16h bfrag = __builtin_shufflevector(blo, bhi, 0,1,2,3,4,5,6,7,8,9,10,11,12,13,14,15);
      acc = __builtin_amdgcn_wmma_f32_16x16x32_f16(false, afrag[i], false, bfrag,
                                                   (short)0, acc, false, false);
    }
    if (N == NPAD || ncol < N) {
      #pragma unroll
      for (int v = 0; v < 8; ++v) {
        int row = row0 + v + ((lane >> 4) << 3);     // C/D layout: VGPR v -> M=v / v+8
        float val = acc[v];
        if constexpr (MODE == 0) {
          out[(size_t)row * NPAD + ncol] = val;
        } else if constexpr (MODE == 1) {
          int bb = row >> 14, l = row & (LL_ - 1);
          out[(size_t)row * 64 + ncol] =
              aux[((size_t)(bb * 64 + ncol) << 14) + l] + val;
        } else {
          int bb = row >> 14, l = row & (LL_ - 1);
          dout[((size_t)(bb * 64 + ncol) << 14) + l] =
              aux[(size_t)row * 64 + ncol] + val;
        }
      }
    }
  }
}

// ---------------- causal depthwise conv1d + bias + SiLU ----------------
__global__ __launch_bounds__(256)
void k_conv_silu(const float* __restrict__ xz, const float* __restrict__ cw,
                 const float* __restrict__ cb,
                 float* __restrict__ xm32, _Float16* __restrict__ xm16) {
  int idx = blockIdx.x * 256 + threadIdx.x;
  if (idx >= MM_ * D_INNER) return;
  int r = idx >> 7, d = idx & 127;
  int bb = r >> 14, l = r & (LL_ - 1);
  float acc = cb[d];
  #pragma unroll
  for (int j = 0; j < D_CONVK; ++j) {
    int ll = l - (D_CONVK - 1) + j;
    if (ll >= 0)
      acc += cw[d * D_CONVK + j] * xz[(((size_t)bb << 14) + ll) * 256 + d];
  }
  float s = acc / (1.f + __expf(-acc));                 // silu
  xm32[idx] = s;
  xm16[idx] = (_Float16)s;
}

// ---------------- selective scan: 2 channels per wave, state n = lane&15 ----------------
__global__ __launch_bounds__(256)
void k_scan(const float* __restrict__ xdb, const float* __restrict__ xm32,
            const float* __restrict__ xz,
            const float* __restrict__ dtw, const float* __restrict__ dtb,
            const float* __restrict__ A_log, const float* __restrict__ Dpv,
            _Float16* __restrict__ y16) {
  int lane = threadIdx.x & 31;
  int wid  = threadIdx.x >> 5;
  int ch   = blockIdx.x * 16 + wid * 2 + (lane >> 4);   // 0..255 = b*128+d
  int bb = ch >> 7, d = ch & 127;
  int n  = lane & 15;
  float Adn  = -__expf(A_log[d * D_STATE + n]);
  float w0 = dtw[0 * D_INNER + d], w1 = dtw[1 * D_INNER + d];
  float w2 = dtw[2 * D_INNER + d], w3 = dtw[3 * D_INNER + d];
  float bias = dtb[d], Dpd = Dpv[d];
  float h = 0.f;
  for (int l = 0; l < LL_; ++l) {
    size_t r = ((size_t)bb << 14) + l;
    const float* xr = xdb + r * XDB_LD;
    float draw  = bias + xr[0]*w0 + xr[1]*w1 + xr[2]*w2 + xr[3]*w3;
    float delta = (draw > 20.f) ? draw : log1pf(__expf(draw));   // softplus
    float xmv = xm32[r * D_INNER + d];
    float Bn  = xr[DT_RANK + n];
    float Cn  = xr[DT_RANK + D_STATE + n];
    h = h * __expf(delta * Adn) + delta * Bn * xmv;
    float p = h * Cn;                                   // reduce over 16 states
    p += __shfl_xor(p, 8, 16);
    p += __shfl_xor(p, 4, 16);
    p += __shfl_xor(p, 2, 16);
    p += __shfl_xor(p, 1, 16);
    if (n == 0) {
      float y  = p + Dpd * xmv;
      float zv = xz[r * 256 + D_INNER + d];
      float yz = y * (zv / (1.f + __expf(-zv)));        // y * silu(z)
      y16[r * D_INNER + d] = (_Float16)yz;
    }
  }
}

// ---------------- depthwise 3x3 (zero pad) + exact GELU gate ----------------
__global__ __launch_bounds__(256)
void k_dwconv_gelu(const float* __restrict__ u, const float* __restrict__ gw,
                   _Float16* __restrict__ v16) {
  int idx = blockIdx.x * 256 + threadIdx.x;
  if (idx >= MM_ * HIDDEN) return;
  int r = idx >> 7, o = idx & 127;
  int bb = r >> 14, hw = r & (LL_ - 1);
  int h = hw >> 7, w = hw & 127;
  float a1 = 0.f, a2 = 0.f;
  #pragma unroll
  for (int dy = 0; dy < 3; ++dy) {
    int hh2 = h + dy - 1;
    if (hh2 < 0 || hh2 >= HH_) continue;
    #pragma unroll
    for (int dx = 0; dx < 3; ++dx) {
      int ww2 = w + dx - 1;
      if (ww2 < 0 || ww2 >= WW_) continue;
      size_t rr = ((size_t)bb << 14) + (hh2 << 7) + ww2;
      a1 += gw[o * 9 + dy * 3 + dx]          * u[rr * 256 + o];
      a2 += gw[(o + 128) * 9 + dy * 3 + dx]  * u[rr * 256 + o + 128];
    }
  }
  float g = 0.5f * a1 * (1.f + erff(a1 * 0.70710678118654752f));
  v16[idx] = (_Float16)(g * a2);
}

extern "C" void kernel_launch(void* const* d_in, const int* in_sizes, int n_in,
                              void* d_out, int out_size, void* d_ws, size_t ws_size,
                              hipStream_t stream) {
  (void)in_sizes; (void)n_in; (void)out_size; (void)ws_size;
  const float* x        = (const float*)d_in[0];
  const float* ln1_w    = (const float*)d_in[1];
  const float* ln1_b    = (const float*)d_in[2];
  const float* in_proj  = (const float*)d_in[3];
  const float* conv_w   = (const float*)d_in[4];
  const float* conv_b   = (const float*)d_in[5];
  const float* x_proj   = (const float*)d_in[6];
  const float* dt_w     = (const float*)d_in[7];
  const float* dt_b     = (const float*)d_in[8];
  const float* A_log    = (const float*)d_in[9];
  const float* Dpv      = (const float*)d_in[10];
  const float* out_proj = (const float*)d_in[11];
  const float* ln2_w    = (const float*)d_in[12];
  const float* ln2_b    = (const float*)d_in[13];
  const float* gin_w    = (const float*)d_in[14];
  const float* gdw_w    = (const float*)d_in[15];
  const float* gout_w   = (const float*)d_in[16];
  float* out = (float*)d_out;

  char* ws = (char*)d_ws;
  _Float16* t16  = (_Float16*)(ws + ( 0ull << 20));  //  4 MB  (ln1 / ln2 output, reused)
  float*    xz   = (float*)   (ws + ( 4ull << 20));  // 32 MB  (in_proj out; reused as u)
  float*    xm32 = (float*)   (ws + (36ull << 20));  // 16 MB
  _Float16* xm16 = (_Float16*)(ws + (52ull << 20));  //  8 MB
  float*    xdb  = (float*)   (ws + (60ull << 20));  //  6 MB  (M x 48)
  _Float16* y16  = (_Float16*)(ws + (66ull << 20));  //  8 MB
  float*    x2   = (float*)   (ws + (74ull << 20));  //  8 MB  (residual, row-major M x 64)
  _Float16* vv16 = (_Float16*)(ws + (82ull << 20));  //  8 MB
  float*    u    = xz;                               // reuse: z consumed before gdfn_in

  // 1) ln1 over channels (NCHW gather) -> f16 rows
  k_layernorm_f16<<<MM_/256, 256, 0, stream>>>(x, 1, ln1_w, ln1_b, t16);
  // 2) in_proj: (M x 64) @ (64 x 256) -> xz
  k_wmma_gemm<64, 256, 256, false, 0><<<MM_/128, 256, 0, stream>>>(
      t16, in_proj, xz, nullptr, nullptr);
  // 3) causal depthwise conv1d + SiLU -> xm
  k_conv_silu<<<(MM_*D_INNER)/256, 256, 0, stream>>>(xz, conv_w, conv_b, xm32, xm16);
  // 4) x_proj: (M x 128) @ (128 x 36) -> xdb (padded stride 48)
  k_wmma_gemm<128, 36, 48, false, 0><<<MM_/128, 256, 0, stream>>>(
      xm16, x_proj, xdb, nullptr, nullptr);
  // 5) selective scan (fused dt_proj + softplus + SSM + C-contraction + silu(z) gate)
  k_scan<<<16, 256, 0, stream>>>(xdb, xm32, xz, dt_w, dt_b, A_log, Dpv, y16);
  // 6) out_proj + residual with original x -> x2 (row-major)
  k_wmma_gemm<128, 64, 64, false, 1><<<MM_/128, 256, 0, stream>>>(
      y16, out_proj, x2, x, nullptr);
  // 7) ln2 (row-major) -> t16
  k_layernorm_f16<<<MM_/256, 256, 0, stream>>>(x2, 0, ln2_w, ln2_b, t16);
  // 8) gdfn_in: (M x 64) @ W^T where W is (256,64) already N x K
  k_wmma_gemm<64, 256, 256, true, 0><<<MM_/128, 256, 0, stream>>>(
      t16, gin_w, u, nullptr, nullptr);
  // 9) depthwise 3x3 + gelu gate -> v
  k_dwconv_gelu<<<(MM_*HIDDEN)/256, 256, 0, stream>>>(u, gdw_w, vv16);
  // 10) gdfn_out (W is (64,128) = N x K) + residual x2, scatter to NCHW d_out
  k_wmma_gemm<128, 64, 64, true, 2><<<MM_/128, 256, 0, stream>>>(
      vv16, gout_w, out, x2, out);
}
